// FilterbankAttention_70145405878564
// MI455X (gfx1250) — matrix operand order repaired and checked by reading, not compiled, not run on hardware
//
#include <hip/hip_runtime.h>
#include <hip/hip_bf16.h>

// ---------------------------------------------------------------------------
// FilterbankAttention (DRAW read) for gfx1250 / MI455X
//   OUT[b] = gamma * F_Y (64x128) @ X[b] (128x128) @ F_X^T (128x64)
// Memory-bound (~320MB @ 23.3TB/s ~= 14us floor); matmuls on bf16 WMMA.
// ---------------------------------------------------------------------------

typedef __attribute__((ext_vector_type(16))) __bf16 v16bf;
typedef __attribute__((ext_vector_type(8)))  float  v8f;
typedef __attribute__((ext_vector_type(4)))  unsigned int v4u;
typedef __attribute__((ext_vector_type(4)))  float  v4f;

union Frag {
    v4u            q[2];
    v16bf          v;
    __bf16         h[16];
    unsigned short s[16];
};
static_assert(sizeof(Frag) == 32, "frag size");

// Native fptrunc (RNE); compiler picks hw cvt if gfx1250 has it.
__device__ __forceinline__ unsigned short f2bf(float f) {
    return __builtin_bit_cast(unsigned short, (__bf16)f);
}

// ---- WMMA bf16 fragment loaders (ISA 7.12.2 layouts, wave32) ---------------
// A (16x32, MxK): lane m=lane&15; lanes16-31 shift K by +8.
//   slots 0..7  -> K = kcol + s + hi*8
//   slots 8..15 -> K = kcol + 16 + (s-8) + hi*8
// => two contiguous 16B LDS loads per lane.
__device__ __forceinline__ v16bf lds_frag_a(const unsigned short* base, int ld,
                                            int mrow, int kcol, int lane) {
    int m  = lane & 15;
    int hi = lane >> 4;
    const unsigned short* p = base + (mrow + m) * ld + kcol + hi * 8;
    Frag f;
    f.q[0] = *(const v4u*)(p);
    f.q[1] = *(const v4u*)(p + 16);
    return f.v;
}

// B (32x16, KxN): lane holds column N = ncol + (lane&15); K as for A (table order).
__device__ __forceinline__ v16bf lds_frag_b(const unsigned short* base, int ld,
                                            int krow, int ncol, int lane) {
    const unsigned short* p = base + (krow + (lane >> 4) * 8) * ld + ncol + (lane & 15);
    Frag f;
    #pragma unroll
    for (int s = 0; s < 8; ++s) {
        f.s[s]     = p[s * ld];
        f.s[s + 8] = p[(s + 16) * ld];
    }
    return f.v;
}

// ---------------------------------------------------------------------------
// Kernel 1: params -> Gaussian filterbanks (global-sum normalized, bf16)
//   FY[n*128+a]  = gamma * exp(-(a-muY_n)^2/2var) / sum(F_Y)
//   FXT[a*64+n]  =         exp(-(a-muX_n)^2/2var) / sum(F_X)   (K-major for B)
// ---------------------------------------------------------------------------
__global__ __launch_bounds__(256)
void fb_kernel(const float* __restrict__ h, const float* __restrict__ Ww,
               const float* __restrict__ Wb,
               unsigned short* __restrict__ FY, unsigned short* __restrict__ FXT) {
    __shared__ float red[256];
    const int tid = threadIdx.x;

    float p[5];
    for (int j = 0; j < 5; ++j) {
        float part = 0.f;
        for (int i = tid; i < 512; i += 256) part += h[i] * Ww[j * 512 + i];
        red[tid] = part; __syncthreads();
        for (int s = 128; s > 0; s >>= 1) {
            if (tid < s) red[tid] += red[tid + s];
            __syncthreads();
        }
        p[j] = red[0] + Wb[j];
        __syncthreads();
    }

    const float var   = __expf(p[2] + 1e-8f);
    const float dd    = __expf(p[3]) * (127.0f / 63.0f);
    const float gX    = 129.0f * (p[0] + 1.0f) * 0.5f;
    const float gY    = 129.0f * (p[1] + 1.0f) * 0.5f;
    const float gamma = __expf(p[4]);
    const float inv2v = 0.5f / var;

    // global sums
    float sx = 0.f, sy = 0.f;
    for (int idx = tid; idx < 64 * 128; idx += 256) {
        int n = idx >> 7, a = idx & 127;
        float off = ((float)n - 32.5f) * dd;
        float tX = (float)a - (gX + off);
        float tY = (float)a - (gY + off);
        sx += __expf(-tX * tX * inv2v);
        sy += __expf(-tY * tY * inv2v);
    }
    red[tid] = sx; __syncthreads();
    for (int s = 128; s > 0; s >>= 1) { if (tid < s) red[tid] += red[tid + s]; __syncthreads(); }
    const float sumX = red[0]; __syncthreads();
    red[tid] = sy; __syncthreads();
    for (int s = 128; s > 0; s >>= 1) { if (tid < s) red[tid] += red[tid + s]; __syncthreads(); }
    const float sumY = red[0]; __syncthreads();

    const float scX = 1.0f / sumX;
    const float scY = gamma / sumY;        // fold gamma into F_Y once

    for (int idx = tid; idx < 64 * 128; idx += 256) {
        int n = idx >> 7, a = idx & 127;
        float off = ((float)n - 32.5f) * dd;
        float tX = (float)a - (gX + off);
        float tY = (float)a - (gY + off);
        FXT[a * 64 + n]  = f2bf(__expf(-tX * tX * inv2v) * scX);
        FY [n * 128 + a] = f2bf(__expf(-tY * tY * inv2v) * scY);
    }
}

// ---------------------------------------------------------------------------
// Kernel 2: one workgroup (8 waves) per batch element; both images.
//   Stage 1: T = FY @ X   (X B-fragments straight from global, read-once)
//   Stage 2: OUT = T @ FXT
// ---------------------------------------------------------------------------
__global__ __launch_bounds__(256)
void glimpse_kernel(const float* __restrict__ x, const float* __restrict__ xh,
                    const unsigned short* __restrict__ gFY,
                    const unsigned short* __restrict__ gFXT,
                    float* __restrict__ out) {
    __shared__ unsigned short sFY [64 * 128];   // 16 KB
    __shared__ unsigned short sFXT[128 * 64];   // 16 KB
    __shared__ unsigned short sT  [64 * 128];   // 16 KB

    const int tid  = threadIdx.x;
    const int b    = blockIdx.x;
    const int lane = tid & 31;
    const int wv   = tid >> 5;        // 8 waves
    const int hi   = lane >> 4;
    const int lm   = lane & 15;

    {   // stage the (tiny, batch-invariant) filterbanks into LDS
        const unsigned int* s1 = (const unsigned int*)gFY;
        const unsigned int* s2 = (const unsigned int*)gFXT;
        unsigned int* d1 = (unsigned int*)sFY;
        unsigned int* d2 = (unsigned int*)sFXT;
        for (int i = tid; i < 4096; i += 256) d1[i] = s1[i];
        for (int i = tid; i < 4096; i += 256) d2[i] = s2[i];
    }

    // Warm L2/L0 for the second image while image 0 is processed
    // (64KB/image; each thread touches its own future fragment lines).
    {
        const float* pre = xh + (size_t)b * (128 * 128);
        #pragma unroll
        for (int kc = 0; kc < 4; ++kc)
            __builtin_prefetch(pre + (size_t)(kc * 32 + lane) * 128 + 16 * wv, 0, 3);
    }
    __syncthreads();

    for (int img = 0; img < 2; ++img) {
        const float* __restrict__ src = (img ? xh : x) + (size_t)b * (128 * 128);

        // ---- Stage 1: wave wv owns T columns [16*wv, 16*wv+16) --------------
        v8f acc[4];
        acc[0] = {}; acc[1] = {}; acc[2] = {}; acc[3] = {};
        #pragma unroll
        for (int kc = 0; kc < 4; ++kc) {
            // X B-fragment: lane reads column (16*wv+lm), K rows in table order
            const float* p = src + (size_t)(kc * 32 + hi * 8) * 128 + 16 * wv + lm;
            Frag bx;
            #pragma unroll
            for (int i = 0; i < 8; ++i) {
                bx.h[i]     = (__bf16)p[i * 128];
                bx.h[i + 8] = (__bf16)p[(i + 16) * 128];
            }
            #pragma unroll
            for (int mt = 0; mt < 4; ++mt) {
                v16bf af = lds_frag_a(sFY, 128, 16 * mt, 32 * kc, lane);
                acc[mt] = __builtin_amdgcn_wmma_f32_16x16x32_bf16(
                    false, af, false, bx.v, (short)0, acc[mt], false, false);
            }
        }
        // T tiles -> LDS (bf16, row-major 64x128); C layout: VGPR r ->
        // row = r + 8*hi, col = lane&15
        #pragma unroll
        for (int mt = 0; mt < 4; ++mt) {
            #pragma unroll
            for (int r = 0; r < 8; ++r) {
                sT[(16 * mt + r + hi * 8) * 128 + 16 * wv + lm] = f2bf(acc[mt][r]);
            }
        }
        __syncthreads();

        // ---- Stage 2: OUT = T @ FXT; wave wv owns tiles 2wv, 2wv+1 ----------
        #pragma unroll
        for (int ti = 0; ti < 2; ++ti) {
            const int t  = 2 * wv + ti;
            const int mt = t >> 2;
            const int nt = t & 3;
            v8f c = {};
            #pragma unroll
            for (int kc = 0; kc < 4; ++kc) {
                v16bf at = lds_frag_a(sT, 128, 16 * mt, 32 * kc, lane);
                v16bf bf = lds_frag_b(sFXT, 64, 32 * kc, 16 * nt, lane);
                c = __builtin_amdgcn_wmma_f32_16x16x32_bf16(
                    false, at, false, bf, (short)0, c, false, false);
            }
            float* o = out + (size_t)b * 8192 + img * 4096
                     + (16 * mt + hi * 8) * 64 + 16 * nt + lm;
            #pragma unroll
            for (int r = 0; r < 8; ++r) o[r * 64] = c[r];
        }
        __syncthreads();   // sT is rewritten by next image's stage 1
    }
}

// ---------------------------------------------------------------------------
extern "C" void kernel_launch(void* const* d_in, const int* in_sizes, int n_in,
                              void* d_out, int out_size, void* d_ws, size_t ws_size,
                              hipStream_t stream) {
    const float* x   = (const float*)d_in[0];
    const float* xh  = (const float*)d_in[1];
    const float* h   = (const float*)d_in[2];
    const float* Ww  = (const float*)d_in[3];
    const float* Wb  = (const float*)d_in[4];
    float* out       = (float*)d_out;

    unsigned short* FY  = (unsigned short*)d_ws;          //  64*128 bf16
    unsigned short* FXT = FY + 64 * 128;                  // 128*64  bf16

    const int batch = in_sizes[0] / (128 * 128);

    fb_kernel<<<1, 256, 0, stream>>>(h, Ww, Wb, FY, FXT);
    glimpse_kernel<<<batch, 256, 0, stream>>>(x, xh, FY, FXT, out);
}
